// TNLayer_11510512353564
// MI455X (gfx1250) — matrix operand order repaired and checked by reading, not compiled, not run on hardware
//
#include <hip/hip_runtime.h>

#define RANK   784
#define DIMF   2
#define BOND   32
#define NLABEL 10
#define BATCH  8192
#define HALF   391          // middle blocks per side

typedef __attribute__((ext_vector_type(16))) _Float16 v16h;
typedef __attribute__((ext_vector_type(8)))  float    v8f;
typedef __attribute__((ext_vector_type(2)))  __fp16   v2fp16;

union U2H { unsigned u; _Float16 h[2]; };
union UPK { v2fp16 v; unsigned u; };
union UB  { v16h v; unsigned u[8]; };

// one v_cvt_pk_rtz_f16_f32
static __device__ __forceinline__ unsigned pk2(float a, float b) {
    UPK x; x.v = __builtin_amdgcn_cvt_pkrtz(a, b); return x.u;
}

// lane^16 exchange: ds_swizzle SWAPX16 (group-of-32: xor=0x10, or=0, and=0x1f)
static __device__ __forceinline__ unsigned swap16(unsigned x) {
    return (unsigned)__builtin_amdgcn_ds_swizzle((int)x, 0x401F);
}

static __device__ __forceinline__ v8f wmma16x16x32(v16h a, v16h b, v8f c) {
    // (neg_a, A, neg_b, B, c_mod, C, reuse_a, reuse_b)
    return __builtin_amdgcn_wmma_f32_16x16x32_f16(false, a, false, b, (short)0, c,
                                                  false, false);
}

// ---------------------------------------------------------------------------
// Kernel 1: swizzle f32 middle weights into f16 WMMA A-matrix layout.
// Tile order per step: mat = g*2 + ct, where g=0 -> W_{d=0} slice and
// g=1 -> (W_{d=1} - W_{d=0}) difference slice (so the chain kernel only needs
// one f1-scaled state copy: v' = W0 x v + (W1-W0) x (f1*v)).
// A k order chosen here (B uses the same): lane L -> row m = L%16;  half h:
//   K = h + 16*(L/16).
// Left uses transposed weights A[c,a] = W[a,d,c]; right uses A[c,e] = W[c,d,e]
// with step order reversed to match the scan direction.
// ---------------------------------------------------------------------------
__global__ void tn_prep_weights(const float* __restrict__ wmL,
                                const float* __restrict__ wmR,
                                _Float16* __restrict__ wsL,
                                _Float16* __restrict__ wsR) {
    const int total = HALF * 4 * 32 * 16;
    int i = blockIdx.x * blockDim.x + threadIdx.x;
    if (i >= total) return;
    int h    = i & 15;
    int lane = (i >> 4) & 31;
    int mat  = (i >> 9) & 3;
    int t    = i >> 11;

    int m = lane & 15;
    int j = h >> 1, p = h & 1;
    int K = (2 * j + p) + 16 * (lane >> 4);
    int g  = mat >> 1;      // 0: W0 slice, 1: W1-W0 slice
    int ct = mat & 1;
    int c  = ct * 16 + m;

    // left: A[m=c_local, K=a] from w_mid_left[t, a, d, c]
    {
        size_t i0 = (((size_t)t * BOND + K) * DIMF + 0) * BOND + c;
        size_t i1 = (((size_t)t * BOND + K) * DIMF + 1) * BOND + c;
        float w = g ? (wmL[i1] - wmL[i0]) : wmL[i0];
        wsL[i] = (_Float16)w;
    }
    // right scan step t uses w_mid_right[HALF-1-t]; A[m=c_local, K=e] = W[c,d,e]
    {
        int tr = HALF - 1 - t;
        size_t i0 = (((size_t)tr * BOND + c) * DIMF + 0) * BOND + K;
        size_t i1 = (((size_t)tr * BOND + c) * DIMF + 1) * BOND + K;
        float w = g ? (wmR[i1] - wmR[i0]) : wmR[i0];
        wsR[i] = (_Float16)w;
    }
}

// ---------------------------------------------------------------------------
// Build the initial B-operand state for one 16-sample tile.
// ---------------------------------------------------------------------------
static __device__ __forceinline__ v16h tn_init_state(const float* __restrict__ inputs,
                                                     const float* __restrict__ w_first,
                                                     const float* __restrict__ w_last,
                                                     bool right, int bcol, int kbase) {
    UB sb;
    float p  = inputs[(size_t)bcol * RANK + (right ? (RANK - 1) : 0)];
    float f0 = 1.0f - p, f1 = p;
#pragma unroll
    for (int j = 0; j < 8; ++j) {
        int k0 = kbase + 2 * j, k1 = k0 + 1;
        float x0, x1;
        if (!right) {   // v0[b,a] = f0*wf[0,a] + f1*wf[1,a]
            x0 = f0 * w_first[k0] + f1 * w_first[BOND + k0];
            x1 = f0 * w_first[k1] + f1 * w_first[BOND + k1];
        } else {        // r0[b,c] = f0*wl[c,0] + f1*wl[c,1]
            x0 = f0 * w_last[k0 * 2] + f1 * w_last[k0 * 2 + 1];
            x1 = f0 * w_last[k1 * 2] + f1 * w_last[k1 * 2 + 1];
        }
        sb.u[j] = pk2(x0, x1);
    }
    return sb.v;
}

// Repack the two 16x16 D tiles (f32, lane=batch col, vgpr=bond row) into the
// next step's B operand (lane=batch col, halves=K): pack f32 pairs to f16
// dwords, then swap halves with lane^16 (ds_swizzle SWAPX16).
// lower lanes keep bonds 0-7 local + bonds 8-15 from lane^16;
// upper lanes take bonds 16-23 from lane^16 + keep bonds 24-31 local.
static __device__ __forceinline__ v16h tn_repack(v8f D0, v8f D1, bool up) {
    unsigned p0[4], p1[4];
#pragma unroll
    for (int j = 0; j < 4; ++j) {
        p0[j] = pk2(D0[2 * j], D0[2 * j + 1]);
        p1[j] = pk2(D1[2 * j], D1[2 * j + 1]);
    }
    UB nb;
#pragma unroll
    for (int j = 0; j < 4; ++j) {
        unsigned e0 = swap16(p0[j]);
        unsigned e1 = swap16(p1[j]);
        nb.u[j]     = up ? e1 : p0[j];
        nb.u[4 + j] = up ? p1[j] : e0;
    }
    return nb.v;
}

static __device__ __forceinline__ void tn_store_state(float* __restrict__ obuf,
                                                      v16h state, int bcol, int kbase) {
    UB fb; fb.v = state;
#pragma unroll
    for (int q = 0; q < 4; ++q) {
        U2H a_, b_;
        a_.u = fb.u[2 * q];
        b_.u = fb.u[2 * q + 1];
        float4 f4 = make_float4((float)a_.h[0], (float)a_.h[1],
                                (float)b_.h[0], (float)b_.h[1]);
        *(float4*)(obuf + (size_t)bcol * BOND + kbase + 4 * q) = f4;
    }
}

// ---------------------------------------------------------------------------
// Kernel 2: each wave runs TWO independent 16-sample chains (same side, same
// step); shared A weight tiles serve 8 WMMAs per 4KB load. Update rule:
//   v' = W0 x v + (W1-W0) x (f1*v)
// so only one per-lane pk_mul_f16 scaled copy is needed and the two slices
// accumulate through the WMMA C operand (no f32 combine VALU).
// blockIdx.x < 32 : left chains, else right. 8 waves/block.
// ---------------------------------------------------------------------------
__global__ void __launch_bounds__(256)
tn_chain(const float* __restrict__ inputs,
         const float* __restrict__ w_first,
         const float* __restrict__ w_last,
         const _Float16* __restrict__ wsL,
         const _Float16* __restrict__ wsR,
         float* __restrict__ vLbuf,
         float* __restrict__ rRbuf) {
    const int lane   = threadIdx.x & 31;
    const int waveId = threadIdx.x >> 5;
    const bool right = (blockIdx.x >= 32);
    const int pair   = (blockIdx.x & 31) * 8 + waveId;   // 0..255
    const int b0     = pair * 32;                        // chain 0 tile base
    const int bcol0  = b0 + (lane & 15);
    const int bcol1  = bcol0 + 16;                       // chain 1 tile base
    const bool up    = (lane >= 16);
    const int kbase  = up ? 16 : 0;

    v16h s0 = tn_init_state(inputs, w_first, w_last, right, bcol0, kbase);
    v16h s1 = tn_init_state(inputs, w_first, w_last, right, bcol1, kbase);

    // f1(site(0))-scaled copies (operand of the difference-slice WMMA)
    float pa = inputs[(size_t)bcol0 * RANK + (right ? (RANK - 2) : 1)];
    float pb = inputs[(size_t)bcol1 * RANK + (right ? (RANK - 2) : 1)];
    v16h s0F = s0 * (_Float16)pa;
    v16h s1F = s1 * (_Float16)pb;

    const v16h* __restrict__ wA = (const v16h*)(right ? wsR : wsL);

    for (int t = 0; t < HALF; ++t) {
        const v16h* base = wA + (size_t)t * 4 * 32;
        // prefetch next step's 4KB of tiles (global_prefetch_b8)
        if (t + 1 < HALF)
            __builtin_prefetch((const char*)(base + 4 * 32) + lane * 128, 0, 0);

        v16h a0 = base[0 * 32 + lane];   // W0 slice, out-bond half 0
        v16h a1 = base[1 * 32 + lane];   // W0 slice, out-bond half 1
        v16h a2 = base[2 * 32 + lane];   // (W1-W0) slice, half 0
        v16h a3 = base[3 * 32 + lane];   // (W1-W0) slice, half 1

        v8f z = {0.f, 0.f, 0.f, 0.f, 0.f, 0.f, 0.f, 0.f};
        // chain 0: two out-bond tiles, slices chained through C
        v8f D00 = wmma16x16x32(a2, s0F, wmma16x16x32(a0, s0, z));
        v8f D01 = wmma16x16x32(a3, s0F, wmma16x16x32(a1, s0, z));
        // chain 1
        v8f D10 = wmma16x16x32(a2, s1F, wmma16x16x32(a0, s1, z));
        v8f D11 = wmma16x16x32(a3, s1F, wmma16x16x32(a1, s1, z));

        s0 = tn_repack(D00, D01, up);
        s1 = tn_repack(D10, D11, up);

        if (t + 1 < HALF) {
            int site = right ? (RANK - 2 - (t + 1)) : (1 + (t + 1));
            float qa = inputs[(size_t)bcol0 * RANK + site];
            float qb = inputs[(size_t)bcol1 * RANK + site];
            s0F = s0 * (_Float16)qa;
            s1F = s1 * (_Float16)qb;
        }
    }

    float* obuf = right ? rRbuf : vLbuf;
    tn_store_state(obuf, s0, bcol0, kbase);
    tn_store_state(obuf, s1, bcol1, kbase);
}

// ---------------------------------------------------------------------------
// Kernel 3: out[b,l] = sum_{a,c} vL[b,a] * w_label[a,l,c] * rR[b,c], minus
// the per-row max. One wave per sample, lane = bond index a.
// ---------------------------------------------------------------------------
__global__ void __launch_bounds__(256)
tn_label(const float* __restrict__ vL,
         const float* __restrict__ rR,
         const float* __restrict__ w_label,
         float* __restrict__ out) {
    const int lane = threadIdx.x & 31;
    const int wave = threadIdx.x >> 5;
    const int b    = blockIdx.x * 8 + wave;

    float va = vL[(size_t)b * BOND + lane];
    float rr = rR[(size_t)b * BOND + lane];
    float rc[32];
#pragma unroll
    for (int c = 0; c < 32; ++c) rc[c] = __shfl(rr, c, 32);

    float acc[NLABEL];
#pragma unroll
    for (int l = 0; l < NLABEL; ++l) {
        float s = 0.f;
#pragma unroll
        for (int c = 0; c < 32; ++c)
            s = fmaf(w_label[((size_t)lane * NLABEL + l) * BOND + c], rc[c], s);
        acc[l] = va * s;
    }
#pragma unroll
    for (int off = 16; off >= 1; off >>= 1)
#pragma unroll
        for (int l = 0; l < NLABEL; ++l)
            acc[l] += __shfl_xor(acc[l], off, 32);

    float m = acc[0];
#pragma unroll
    for (int l = 1; l < NLABEL; ++l) m = fmaxf(m, acc[l]);
    if (lane < NLABEL) out[(size_t)b * NLABEL + lane] = acc[lane] - m;
}

// ---------------------------------------------------------------------------
extern "C" void kernel_launch(void* const* d_in, const int* in_sizes, int n_in,
                              void* d_out, int out_size, void* d_ws, size_t ws_size,
                              hipStream_t stream) {
    const float* inputs      = (const float*)d_in[0];
    const float* w_first     = (const float*)d_in[1];
    const float* w_mid_left  = (const float*)d_in[2];
    const float* w_label     = (const float*)d_in[3];
    const float* w_mid_right = (const float*)d_in[4];
    const float* w_last      = (const float*)d_in[5];
    float* out = (float*)d_out;

    char* ws = (char*)d_ws;
    const size_t sideBytes = (size_t)HALF * 4 * 32 * 16 * sizeof(_Float16); // 1,601,536 (256B-mult)
    _Float16* wsL = (_Float16*)ws;
    _Float16* wsR = (_Float16*)(ws + sideBytes);
    float* vLbuf  = (float*)(ws + 2 * sideBytes);
    float* rRbuf  = vLbuf + (size_t)BATCH * BOND;

    (void)in_sizes; (void)n_in; (void)out_size; (void)ws_size;

    const int prepTotal = HALF * 4 * 32 * 16;
    tn_prep_weights<<<(prepTotal + 255) / 256, 256, 0, stream>>>(
        w_mid_left, w_mid_right, wsL, wsR);

    tn_chain<<<64, 256, 0, stream>>>(inputs, w_first, w_last, wsL, wsR,
                                     vLbuf, rRbuf);

    tn_label<<<BATCH / 8, 256, 0, stream>>>(vLbuf, rRbuf, w_label, out);
}
